// AlexNet_51539607552765
// MI455X (gfx1250) — matrix operand order repaired and checked by reading, not compiled, not run on hardware
//
#include <hip/hip_runtime.h>
#include <hip/hip_bf16.h>

typedef _Float16 v16h __attribute__((ext_vector_type(16)));
typedef _Float16 v8h  __attribute__((ext_vector_type(8)));
typedef float    v8f  __attribute__((ext_vector_type(8)));

#define WAVES_PER_BLOCK 8
#define MAX_KP 3456   // largest padded K among all conv layers (conv4/conv5)

// ---------------------------------------------------------------------------
// Weight pack: f32 [M,K] -> f16 [Mp,Kp], zero padded. One thread per dst elem.
// ---------------------------------------------------------------------------
__global__ void pack_f16(const float* __restrict__ src, _Float16* __restrict__ dst,
                         int M, int K, int Mp, int Kp)
{
    long long i = (long long)blockIdx.x * blockDim.x + threadIdx.x;
    long long total = (long long)Mp * Kp;
    if (i >= total) return;
    int k = (int)(i % Kp);
    int m = (int)(i / Kp);
    float v = (m < M && k < K) ? src[(long long)m * K + k] : 0.0f;
    dst[i] = (_Float16)v;
}

// ---------------------------------------------------------------------------
// Implicit-GEMM convolution, fused bias + ReLU, f16 WMMA / f32 accumulate.
//   M = Cout (mult of 16), N = B*OH*OW, K = Cin*KH*KW (weights padded to Kp%32==0)
//   One wave computes a 16(M) x 32(N) tile: two v_wmma sharing the A fragment.
//   LDS LUT removes divisions; 32-bit indices (all tensors < 2^31 elements);
//   two-phase gather (loads first, converts second) so loads can clause.
// ---------------------------------------------------------------------------
__global__ __launch_bounds__(256) void conv_igemm_wmma(
    const float* __restrict__ x, const _Float16* __restrict__ wp,
    const float* __restrict__ bias, float* __restrict__ out,
    int Bn, int Cin, int H, int W,
    int Cout, int KH, int KW,
    int OH, int OW, int stride, int pad, int Kp)
{
    __shared__ int2 lut[MAX_KP];

    const int HW  = H * W;
    const int KHW = KH * KW;

    // ---- cooperative LUT fill
    for (int k = threadIdx.x; k < Kp; k += 256) {
        int cin = k / KHW;
        int r   = k - cin * KHW;
        int kh  = r / KW;
        int kw  = r - kh * KW;
        lut[k] = make_int2(cin * HW + kh * W + kw, (kh << 16) | kw);
    }
    __syncthreads();

    const int lane        = threadIdx.x & 31;
    const int waveInBlock = threadIdx.x >> 5;
    const long long waveId = (long long)blockIdx.x * WAVES_PER_BLOCK + waveInBlock;

    const int  OHW  = OH * OW;
    const long long Ntot = (long long)Bn * OHW;
    const int  Mtiles = Cout >> 4;
    const long long Ntiles = (Ntot + 31) >> 5;
    if (waveId >= (long long)Mtiles * Ntiles) return;  // uniform, after barrier

    const int       mTile = (int)(waveId % Mtiles);
    const long long nTile = waveId / Mtiles;

    const int mLane = lane & 15;
    const int blk   = lane >> 4;

    const long long wRow = (long long)(mTile * 16 + mLane) * Kp;

    // ---- column contexts (two N columns per lane), 32-bit bases
    int  cb[2], cpix[2], cih0[2], ciw0[2], cbase[2];
    bool cok[2];
    #pragma unroll
    for (int s = 0; s < 2; ++s) {
        long long nIdx = nTile * 32 + s * 16 + mLane;
        bool ok = (nIdx < Ntot);
        long long nc = ok ? nIdx : (Ntot - 1);
        int b   = (int)(nc / OHW);
        int pix = (int)(nc % OHW);
        int oh  = pix / OW;
        int ow  = pix - oh * OW;
        int ih0 = oh * stride - pad;
        int iw0 = ow * stride - pad;
        cb[s] = b; cpix[s] = pix; cok[s] = ok;
        cih0[s] = ih0; ciw0[s] = iw0;
        cbase[s] = b * Cin * HW + ih0 * W + iw0;   // fits int32
    }
    const int CinHW = Cin * HW;

    v8f acc0 = {}, acc1 = {};
    for (int k0 = 0; k0 < Kp; k0 += 32) {
        __builtin_prefetch(wp + wRow + k0 + 128, 0, 1);   // global_prefetch_b8

        // ---- A fragment: two aligned 16-byte loads (global_load_b128)
        v8h lo = *(const v8h*)(wp + wRow + k0 + blk * 16);
        v8h hi = *(const v8h*)(wp + wRow + k0 + blk * 16 + 8);
        v16h a = __builtin_shufflevector(lo, hi,
                     0, 1, 2, 3, 4, 5, 6, 7, 8, 9, 10, 11, 12, 13, 14, 15);

        // ---- phase 1: issue all 32 gathers (bounds packed into bitmasks)
        const int kb = k0 + blk * 16;
        float r0[16], r1[16];
        unsigned msk0 = 0, msk1 = 0;
        #pragma unroll
        for (int h = 0; h < 16; ++h) {
            int2 e  = lut[kb + h];
            int  kh = e.y >> 16;
            int  kw = e.y & 0xffff;
            {
                int ih = cih0[0] + kh, iw = ciw0[0] + kw;
                int inb = (int)((unsigned)ih < (unsigned)H) &
                          (int)((unsigned)iw < (unsigned)W) &
                          (int)(e.x < CinHW);
                int idx = inb ? (cbase[0] + e.x) : 0;
                r0[h] = x[idx];
                msk0 |= (unsigned)inb << h;
            }
            {
                int ih = cih0[1] + kh, iw = ciw0[1] + kw;
                int inb = (int)((unsigned)ih < (unsigned)H) &
                          (int)((unsigned)iw < (unsigned)W) &
                          (int)(e.x < CinHW);
                int idx = inb ? (cbase[1] + e.x) : 0;
                r1[h] = x[idx];
                msk1 |= (unsigned)inb << h;
            }
        }

        // ---- phase 2: select + convert (single wait covers the load clause)
        v16h bf0, bf1;
        #pragma unroll
        for (int h = 0; h < 16; ++h) {
            float v0 = ((msk0 >> h) & 1u) ? r0[h] : 0.0f;
            float v1 = ((msk1 >> h) & 1u) ? r1[h] : 0.0f;
            bf0[h] = (_Float16)v0;
            bf1[h] = (_Float16)v1;
        }

        acc0 = __builtin_amdgcn_wmma_f32_16x16x32_f16(
                   false, a, false, bf0, (short)0, acc0, false, false);
        acc1 = __builtin_amdgcn_wmma_f32_16x16x32_f16(
                   false, a, false, bf1, (short)0, acc1, false, false);
    }

    // ---- store: lane holds column n; VGPR r -> row m = r + 8*blk (32-bit idx)
    #pragma unroll
    for (int s = 0; s < 2; ++s) {
        if (!cok[s]) continue;
        const v8f& acc = (s == 0) ? acc0 : acc1;
        #pragma unroll
        for (int r8 = 0; r8 < 8; ++r8) {
            int c = mTile * 16 + r8 + 8 * blk;
            float v = acc[r8] + bias[c];
            v = v > 0.0f ? v : 0.0f;
            out[(cb[s] * Cout + c) * OHW + cpix[s]] = v;
        }
    }
}

// ---------------------------------------------------------------------------
// Fully-connected GEMM, fused bias + ReLU.
//   out[n, m] = relu( sum_k in[n,k] * wgt[m,k] + bias[m] )
//   Weights pre-packed f16 [Mp,K], Mp%16==0, K%32==0. 16x32 tile per wave.
//   All offsets fit int32 (max 4096*9216 < 2^31 per row base).
// ---------------------------------------------------------------------------
__global__ __launch_bounds__(256) void fc_wmma(
    const float* __restrict__ in, const _Float16* __restrict__ wp,
    const float* __restrict__ bias, float* __restrict__ out,
    int Bn, int K, int M, int Mp)
{
    const int lane        = threadIdx.x & 31;
    const int waveInBlock = threadIdx.x >> 5;
    const long long waveId = (long long)blockIdx.x * WAVES_PER_BLOCK + waveInBlock;

    const int Mtiles = Mp >> 4;
    const int Ntiles = (Bn + 31) >> 5;
    if (waveId >= (long long)Mtiles * Ntiles) return;

    const int mTile = (int)(waveId % Mtiles);
    const int nTile = (int)(waveId / Mtiles);

    const int mLane = lane & 15;
    const int blk   = lane >> 4;

    const long long wRow = (long long)(mTile * 16 + mLane) * K;

    int nCol[2]; bool nOK[2]; int inRow[2];
    #pragma unroll
    for (int s = 0; s < 2; ++s) {
        int n = nTile * 32 + s * 16 + mLane;
        nOK[s]  = (n < Bn);
        nCol[s] = n;
        inRow[s] = (nOK[s] ? n : 0) * K;
    }

    v8f acc0 = {}, acc1 = {};
    for (int k0 = 0; k0 < K; k0 += 32) {
        __builtin_prefetch(wp + wRow + k0 + 256, 0, 1);

        v8h lo = *(const v8h*)(wp + wRow + k0 + blk * 16);
        v8h hi = *(const v8h*)(wp + wRow + k0 + blk * 16 + 8);
        v16h a = __builtin_shufflevector(lo, hi,
                     0, 1, 2, 3, 4, 5, 6, 7, 8, 9, 10, 11, 12, 13, 14, 15);

        v16h bf0, bf1;
        #pragma unroll
        for (int s = 0; s < 2; ++s) {
            const float4* bp = (const float4*)(in + inRow[s] + k0 + blk * 16);
            float4 q0 = bp[0], q1 = bp[1], q2 = bp[2], q3 = bp[3];
            v16h& bf = (s == 0) ? bf0 : bf1;
            bf[0]  = (_Float16)q0.x; bf[1]  = (_Float16)q0.y;
            bf[2]  = (_Float16)q0.z; bf[3]  = (_Float16)q0.w;
            bf[4]  = (_Float16)q1.x; bf[5]  = (_Float16)q1.y;
            bf[6]  = (_Float16)q1.z; bf[7]  = (_Float16)q1.w;
            bf[8]  = (_Float16)q2.x; bf[9]  = (_Float16)q2.y;
            bf[10] = (_Float16)q2.z; bf[11] = (_Float16)q2.w;
            bf[12] = (_Float16)q3.x; bf[13] = (_Float16)q3.y;
            bf[14] = (_Float16)q3.z; bf[15] = (_Float16)q3.w;
        }

        acc0 = __builtin_amdgcn_wmma_f32_16x16x32_f16(
                   false, a, false, bf0, (short)0, acc0, false, false);
        acc1 = __builtin_amdgcn_wmma_f32_16x16x32_f16(
                   false, a, false, bf1, (short)0, acc1, false, false);
    }

    #pragma unroll
    for (int s = 0; s < 2; ++s) {
        if (!nOK[s]) continue;
        const v8f& acc = (s == 0) ? acc0 : acc1;
        #pragma unroll
        for (int r8 = 0; r8 < 8; ++r8) {
            int c = mTile * 16 + r8 + 8 * blk;
            if (c < M) {
                float v = acc[r8] + bias[c];
                v = v > 0.0f ? v : 0.0f;
                out[nCol[s] * M + c] = v;
            }
        }
    }
}

// ---------------------------------------------------------------------------
// Local response normalization (n=5, k=2, alpha=1e-4, beta=0.75)
// ---------------------------------------------------------------------------
__global__ void lrn_kernel(const float* __restrict__ in, float* __restrict__ out,
                           int Bn, int C, int HW)
{
    long long i = (long long)blockIdx.x * blockDim.x + threadIdx.x;
    long long total = (long long)Bn * C * HW;
    if (i >= total) return;
    int hw = (int)(i % HW);
    long long t = i / HW;
    int c = (int)(t % C);
    int b = (int)(t / C);

    const float* base = in + (long long)b * C * HW + hw;
    int lo = c - 2 < 0 ? 0 : c - 2;
    int hi = c + 2 >= C ? C - 1 : c + 2;
    float sum = 0.0f;
    for (int j = lo; j <= hi; ++j) {
        float v = base[(long long)j * HW];
        sum += v * v;
    }
    float xv = base[(long long)c * HW];
    out[i] = xv * __powf(2.0f + 0.0001f * sum, -0.75f);
}

// ---------------------------------------------------------------------------
// 3x3 stride-2 VALID max-pool
// ---------------------------------------------------------------------------
__global__ void maxpool3x3s2(const float* __restrict__ in, float* __restrict__ out,
                             int Bn, int C, int H, int W, int OH, int OW)
{
    long long i = (long long)blockIdx.x * blockDim.x + threadIdx.x;
    long long total = (long long)Bn * C * OH * OW;
    if (i >= total) return;
    int ow = (int)(i % OW);
    long long t = i / OW;
    int oh = (int)(t % OH); t /= OH;
    int c  = (int)(t % C);
    int b  = (int)(t / C);

    const float* base = in + ((long long)b * C + c) * H * W;
    int h0 = oh * 2, w0 = ow * 2;
    float m = -INFINITY;
    #pragma unroll
    for (int kh = 0; kh < 3; ++kh)
        #pragma unroll
        for (int kw = 0; kw < 3; ++kw)
            m = fmaxf(m, base[(long long)(h0 + kh) * W + (w0 + kw)]);
    out[i] = m;
}

// ---------------------------------------------------------------------------
// Host launcher
// ---------------------------------------------------------------------------
static inline int conv_blocks(int Cout, long long Bn, long long OH, long long OW)
{
    long long Ntot  = Bn * OH * OW;
    long long tiles = (long long)(Cout / 16) * ((Ntot + 31) / 32);
    return (int)((tiles + WAVES_PER_BLOCK - 1) / WAVES_PER_BLOCK);
}

extern "C" void kernel_launch(void* const* d_in, const int* in_sizes, int n_in,
                              void* d_out, int out_size, void* d_ws, size_t ws_size,
                              hipStream_t stream)
{
    const float* x   = (const float*)d_in[0];
    const float* W1  = (const float*)d_in[1];  const float* b1  = (const float*)d_in[2];
    const float* W2  = (const float*)d_in[3];  const float* b2  = (const float*)d_in[4];
    const float* W3  = (const float*)d_in[5];  const float* b3  = (const float*)d_in[6];
    const float* W4  = (const float*)d_in[7];  const float* b4  = (const float*)d_in[8];
    const float* W5  = (const float*)d_in[9];  const float* b5  = (const float*)d_in[10];
    const float* Wf1 = (const float*)d_in[11]; const float* bf1 = (const float*)d_in[12];
    const float* Wf2 = (const float*)d_in[13]; const float* bf2 = (const float*)d_in[14];
    const float* Wf3 = (const float*)d_in[15]; const float* bf3 = (const float*)d_in[16];
    float* out = (float*)d_out;

    const int B = 128;
    const int T = 256;
    char* ws = (char*)d_ws;

    // activation ping-pong (largest live tensor = 148.7 MB)
    float* bufA = (float*)ws;
    float* bufB = (float*)(ws + (size_t)159383552);   // 152 MiB

    // packed-weight arena at +304 MiB (all offsets 256B-aligned)
    size_t wo = (size_t)318767104;
    auto carve = [&](size_t bytes) {
        _Float16* p = (_Float16*)(ws + wo);
        wo += (bytes + 255) & ~(size_t)255;
        return p;
    };
    _Float16* Wp1  = carve((size_t)96   * 384  * 2);
    _Float16* Wp2  = carve((size_t)256  * 2400 * 2);
    _Float16* Wp3  = carve((size_t)384  * 2304 * 2);
    _Float16* Wp4  = carve((size_t)384  * 3456 * 2);
    _Float16* Wp5  = carve((size_t)256  * 3456 * 2);
    _Float16* Wpf1 = carve((size_t)4096 * 9216 * 2);
    _Float16* Wpf2 = carve((size_t)4096 * 4096 * 2);
    _Float16* Wpf3 = carve((size_t)1008 * 4096 * 2);

    auto eltBlocks = [](long long n) { return (int)((n + 255) / 256); };
    auto pack = [&](const float* src, _Float16* dst, int M, int K, int Mp, int Kp) {
        pack_f16<<<eltBlocks((long long)Mp * Kp), T, 0, stream>>>(src, dst, M, K, Mp, Kp);
    };

    // ---- pack all weights to f16 (K padded to %32, fc3 M padded to 1008)
    pack(W1,  Wp1,  96,   363,  96,   384);
    pack(W2,  Wp2,  256,  2400, 256,  2400);
    pack(W3,  Wp3,  384,  2304, 384,  2304);
    pack(W4,  Wp4,  384,  3456, 384,  3456);
    pack(W5,  Wp5,  256,  3456, 256,  3456);
    pack(Wf1, Wpf1, 4096, 9216, 4096, 9216);
    pack(Wf2, Wpf2, 4096, 4096, 4096, 4096);
    pack(Wf3, Wpf3, 1000, 4096, 1008, 4096);

    // conv1: (128,3,224,224) -> (128,96,55,55), s4 p2
    conv_igemm_wmma<<<conv_blocks(96, B, 55, 55), T, 0, stream>>>(
        x, Wp1, b1, bufA, B, 3, 224, 224, 96, 11, 11, 55, 55, 4, 2, 384);
    lrn_kernel<<<eltBlocks((long long)B * 96 * 55 * 55), T, 0, stream>>>(
        bufA, bufB, B, 96, 55 * 55);
    maxpool3x3s2<<<eltBlocks((long long)B * 96 * 27 * 27), T, 0, stream>>>(
        bufB, bufA, B, 96, 55, 55, 27, 27);

    // conv2: (128,96,27,27) -> (128,256,27,27), s1 p2
    conv_igemm_wmma<<<conv_blocks(256, B, 27, 27), T, 0, stream>>>(
        bufA, Wp2, b2, bufB, B, 96, 27, 27, 256, 5, 5, 27, 27, 1, 2, 2400);
    lrn_kernel<<<eltBlocks((long long)B * 256 * 27 * 27), T, 0, stream>>>(
        bufB, bufA, B, 256, 27 * 27);
    maxpool3x3s2<<<eltBlocks((long long)B * 256 * 13 * 13), T, 0, stream>>>(
        bufA, bufB, B, 256, 27, 27, 13, 13);

    // conv3..conv5 (s1 p1, 13x13)
    conv_igemm_wmma<<<conv_blocks(384, B, 13, 13), T, 0, stream>>>(
        bufB, Wp3, b3, bufA, B, 256, 13, 13, 384, 3, 3, 13, 13, 1, 1, 2304);
    conv_igemm_wmma<<<conv_blocks(384, B, 13, 13), T, 0, stream>>>(
        bufA, Wp4, b4, bufB, B, 384, 13, 13, 384, 3, 3, 13, 13, 1, 1, 3456);
    conv_igemm_wmma<<<conv_blocks(256, B, 13, 13), T, 0, stream>>>(
        bufB, Wp5, b5, bufA, B, 384, 13, 13, 256, 3, 3, 13, 13, 1, 1, 3456);
    maxpool3x3s2<<<eltBlocks((long long)B * 256 * 6 * 6), T, 0, stream>>>(
        bufA, bufB, B, 256, 13, 13, 6, 6);

    // fc layers: tiles = (Mp/16) * ceil(B/32)
    {
        long long tiles = (4096 / 16) * ((B + 31) / 32);
        fc_wmma<<<(int)((tiles + 7) / 8), T, 0, stream>>>(bufB, Wpf1, bf1, bufA, B, 9216, 4096, 4096);
    }
    {
        long long tiles = (4096 / 16) * ((B + 31) / 32);
        fc_wmma<<<(int)((tiles + 7) / 8), T, 0, stream>>>(bufA, Wpf2, bf2, bufB, B, 4096, 4096, 4096);
    }
    {
        long long tiles = (1008 / 16) * ((B + 31) / 32);
        fc_wmma<<<(int)((tiles + 7) / 8), T, 0, stream>>>(bufB, Wpf3, bf3, out, B, 4096, 1000, 1008);
    }
}
